// FCPairedLayer_75642964017538
// MI455X (gfx1250) — matrix operand order repaired and checked by reading, not compiled, not run on hardware
//
#include <hip/hip_runtime.h>
#include <hip/hip_bf16.h>

typedef _Float16 half16 __attribute__((ext_vector_type(16)));
typedef _Float16 half8  __attribute__((ext_vector_type(8)));
typedef float    float8 __attribute__((ext_vector_type(8)));

#define Bx 2
#define Nn 1024
#define Cc 128
#define H1 128
#define H2 64

// ---------------------------------------------------------------------------
// Kernel 1: a_pre[b,i,k] = f16( x[b,i,:] @ W1[:C, k] + b1[k] )
//           r_pre[b,j,k] = f16( x[b,j,:] @ W1[C:, k] )
// ---------------------------------------------------------------------------
__global__ void fcpl_precompute_ar(const float* __restrict__ x,
                                   const float* __restrict__ W1,
                                   const float* __restrict__ b1,
                                   _Float16* __restrict__ a_pre,
                                   _Float16* __restrict__ r_pre) {
    const int row = blockIdx.x;          // b*N + i
    const int k   = threadIdx.x;         // 0..127
    __shared__ float xs[Cc];
    xs[k] = x[row * Cc + k];
    __syncthreads();
    float aa = b1[k];
    float rr = 0.0f;
    #pragma unroll 4
    for (int c = 0; c < Cc; ++c) {
        const float xv = xs[c];
        aa = fmaf(xv, W1[c * H1 + k], aa);
        rr = fmaf(xv, W1[(Cc + c) * H1 + k], rr);
    }
    a_pre[row * H1 + k] = (_Float16)aa;
    r_pre[row * H1 + k] = (_Float16)rr;
}

// ---------------------------------------------------------------------------
// Kernel 2: w2t[n*128 + k] = (f16) W2[k*64 + n]   (transpose, k-contiguous)
// ---------------------------------------------------------------------------
__global__ void fcpl_convert_w2(const float* __restrict__ W2,
                                _Float16* __restrict__ w2t) {
    const int idx = blockIdx.x * 256 + threadIdx.x;   // 0..8191
    if (idx < H1 * H2) {
        const int k = idx >> 6;       // 0..127
        const int n = idx & 63;       // 0..63
        w2t[n * H1 + k] = (_Float16)W2[k * H2 + n];
    }
}

// ---------------------------------------------------------------------------
// Kernel 3: transposed pair-MLP via WMMA.
// Block = 512 threads (16 waves) covers 16 i's (one per wave) x 64 j's
// (4 j-tiles). Per wave per j-tile:  h2^T = W2^T x h1^T  with
//   A = W2^T chunk [16 n x 32 k]  (pure LDS loads, constant)
//   B = relu(a_i + r_j) [32 k x 16 pairs]  (pk_add+pk_max, built once)
// 16x v_wmma_f32_16x16x32_f16; D lane = pair, so epilogue relu(.+b2)@W3
// is lane-local float8 math + ONE shfl_xor(16); coalesced 1-float stores.
// ---------------------------------------------------------------------------
#define AR_STR 136  // f16 LDS row stride (136*2B -> 4-bank rotation)
#define W_STR  136

__launch_bounds__(512)
__global__ void fcpl_pair_mlp(const _Float16* __restrict__ a_pre,
                              const _Float16* __restrict__ r_pre,
                              const _Float16* __restrict__ w2t,
                              const float* __restrict__ b2,
                              const float* __restrict__ W3,
                              const float* __restrict__ b3,
                              float* __restrict__ out) {
    __shared__ _Float16 aL[16 * AR_STR];            // 4.25 KB
    __shared__ _Float16 rL[64 * AR_STR];            // 17 KB
    __shared__ _Float16 wL[H2 * W_STR];             // 17 KB
    __shared__ __align__(16) float b2L[H2];         // 256 B
    __shared__ __align__(16) float w3L[H2];         // 256 B

    const int tid = threadIdx.x;
    const int b   = blockIdx.z;
    const int ib  = blockIdx.y << 4;   // i-tile base (16 i's)
    const int jb0 = blockIdx.x << 6;   // j-group base (64 j's)
    float* outb = out + (size_t)b * Nn * Nn;

    // Whole j-group in masked region (all j <= i): zero-fill, skip compute.
    if (jb0 + 63 <= ib) {
        for (int t = tid; t < 16 * 64; t += 512) {
            const int m = t >> 6, j = t & 63;
            outb[(size_t)(ib + m) * Nn + (jb0 + j)] = 0.0f;
        }
        return;
    }

    // ---- stage a (16 rows), r (64 rows), W2^T, b2, W3 into LDS ------------
    {
        const unsigned* ag = (const unsigned*)(a_pre + (((size_t)b * Nn + ib) << 7));
        for (int idx = tid; idx < (16 * H1) / 2; idx += 512) {   // 1024 dwords
            const int row = idx >> 6, kp = idx & 63;
            *(unsigned*)&aL[row * AR_STR + kp * 2] = ag[idx];
        }
        const unsigned* rg = (const unsigned*)(r_pre + (((size_t)b * Nn + jb0) << 7));
        for (int idx = tid; idx < (64 * H1) / 2; idx += 512) {   // 4096 dwords
            const int row = idx >> 6, kp = idx & 63;
            *(unsigned*)&rL[row * AR_STR + kp * 2] = rg[idx];
        }
        const unsigned* wg = (const unsigned*)w2t;
        for (int idx = tid; idx < (H2 * H1) / 2; idx += 512) {   // 4096 dwords
            const int n = idx >> 6, kp = idx & 63;
            *(unsigned*)&wL[n * W_STR + kp * 2] = wg[idx];
        }
        if (tid < H2) { b2L[tid] = b2[tid]; w3L[tid] = W3[tid]; }
    }
    __syncthreads();

    const int wave = tid >> 5;
    const int lane = tid & 31;
    const int lrow = lane & 15;   // B column (pair) / A row (n)
    const int hi   = lane >> 4;
    const int i    = ib + wave;
    const float bb3 = b3[0];

    const _Float16* arow = &aL[wave * AR_STR];
    const half8 h8z = (half8)(_Float16)0.0f;

    for (int jt = 0; jt < 4; ++jt) {                 // block-uniform loop
        const int jb = jb0 + (jt << 4);
        if (jb + 15 <= ib) {                         // fully masked j-tile
            if (tid < 256) {
                const int m = tid >> 4, j = tid & 15;
                outb[(size_t)(ib + m) * Nn + (jb + j)] = 0.0f;
            }
            continue;
        }

        // ---- build B fragments: B[t] = relu(a_i + r_j)[k = kc*32+hi*16+t]
        const _Float16* rrow = &rL[((jt << 4) + lrow) * AR_STR];
        half16 Bfr[4];
        #pragma unroll
        for (int kc = 0; kc < 4; ++kc) {
            const int k0 = kc * 32 + hi * 16;
            union { half16 v; half8 h[2]; } Bu;
            const half8 a0 = *(const half8*)(arow + k0);
            const half8 a1 = *(const half8*)(arow + k0 + 8);
            const half8 r0 = *(const half8*)(rrow + k0);
            const half8 r1 = *(const half8*)(rrow + k0 + 8);
            Bu.h[0] = __builtin_elementwise_max(a0 + r0, h8z);
            Bu.h[1] = __builtin_elementwise_max(a1 + r1, h8z);
            Bfr[kc] = Bu.v;
        }

        // ---- 4 M-chunks of n; A = W2^T from LDS; lane-local epilogue ------
        float8 sv = (float8)0;
        #pragma unroll
        for (int mt = 0; mt < 4; ++mt) {
            const _Float16* wp = &wL[((mt << 4) + lrow) * W_STR];
            float8 acc = (float8)0;
            #pragma unroll
            for (int kc = 0; kc < 4; ++kc) {
                union { half16 v; half8 h[2]; } Au;
                Au.h[0] = *(const half8*)(wp + kc * 32 + hi * 16);
                Au.h[1] = *(const half8*)(wp + kc * 32 + hi * 16 + 8);
                acc = __builtin_amdgcn_wmma_f32_16x16x32_f16(
                    false, Au.v, false, Bfr[kc], (short)0, acc, false, false);
            }
            // acc[p] = h2[pair, n] with n = mt*16 + hi*8 + p
            const float* b2p = &b2L[(mt << 4) + hi * 8];
            const float* w3p = &w3L[(mt << 4) + hi * 8];
            float8 b2v, w3v;
            ((float4*)&b2v)[0] = *(const float4*)(b2p);
            ((float4*)&b2v)[1] = *(const float4*)(b2p + 4);
            ((float4*)&w3v)[0] = *(const float4*)(w3p);
            ((float4*)&w3v)[1] = *(const float4*)(w3p + 4);
            const float8 t = __builtin_elementwise_max(acc + b2v, (float8)0);
            sv += t * w3v;
        }

        // ---- lane-local horizontal sum + one cross-half shuffle -----------
        float s = ((sv[0] + sv[1]) + (sv[2] + sv[3])) +
                  ((sv[4] + sv[5]) + (sv[6] + sv[7]));
        s += __shfl_xor(s, 16, 32);

        if (hi == 0) {                               // 16 coalesced stores
            const int j = jb + lrow;
            outb[(size_t)i * Nn + j] = (j > i) ? (s + bb3) : 0.0f;
        }
    }
}

// ---------------------------------------------------------------------------
extern "C" void kernel_launch(void* const* d_in, const int* in_sizes, int n_in,
                              void* d_out, int out_size, void* d_ws, size_t ws_size,
                              hipStream_t stream) {
    const float* x  = (const float*)d_in[0];
    const float* W1 = (const float*)d_in[1];
    const float* b1 = (const float*)d_in[2];
    const float* W2 = (const float*)d_in[3];
    const float* b2 = (const float*)d_in[4];
    const float* W3 = (const float*)d_in[5];
    const float* b3 = (const float*)d_in[6];
    float* out = (float*)d_out;

    // workspace: a_pre f16 (512KB) | r_pre f16 (512KB) | w2t f16 (16KB)
    _Float16* a_pre = (_Float16*)d_ws;
    _Float16* r_pre = a_pre + (size_t)Bx * Nn * H1;
    _Float16* w2t   = r_pre + (size_t)Bx * Nn * H1;

    fcpl_precompute_ar<<<Bx * Nn, Cc, 0, stream>>>(x, W1, b1, a_pre, r_pre);
    fcpl_convert_w2<<<(H1 * H2 + 255) / 256, 256, 0, stream>>>(W2, w2t);

    dim3 grid(Nn / 64, Nn / 16, Bx);
    fcpl_pair_mlp<<<grid, 512, 0, stream>>>(a_pre, r_pre, w2t, b2, W3, b3, out);
}
